// LengthRegulator_50560355008676
// MI455X (gfx1250) — matrix operand order repaired and checked by reading, not compiled, not run on hardware
//
#include <hip/hip_runtime.h>

#define BB 16
#define SS 1024
#define CC 256
#define FF 256
#define SP 1026   // S + 2 (zero-padded sequence for K=3 'same' conv)

typedef __attribute__((ext_vector_type(16))) _Float16 v16h;
typedef __attribute__((ext_vector_type(8)))  _Float16 v8h;
typedef __attribute__((ext_vector_type(8)))  float    v8f;

// ---------------------------------------------------------------------------
// Pack x (B,S,C) f32 -> Xp (B,SP,C) f16 with zero pad rows at sp==0, sp==S+1
// ---------------------------------------------------------------------------
__global__ __launch_bounds__(256)
void pack_x_kernel(const float* __restrict__ x, _Float16* __restrict__ Xp, int total)
{
    int i = blockIdx.x * 256 + threadIdx.x;
    if (i >= total) return;
    int c  = i & (CC - 1);
    int sp = (i >> 8) % SP;
    int b  = i / (SP * CC);
    float v = 0.0f;
    if (sp >= 1 && sp <= SS)
        v = x[((size_t)b * SS + (sp - 1)) * CC + c];
    Xp[i] = (_Float16)v;
}

// ---------------------------------------------------------------------------
// Pack conv weight (F, C, K=3) f32 -> Wp (3, F, C) f16  (B-operand friendly:
// each lane = one output column f, contiguous K(=c) values)
// ---------------------------------------------------------------------------
__global__ __launch_bounds__(256)
void pack_w_kernel(const float* __restrict__ w, _Float16* __restrict__ Wp, int total)
{
    int i = blockIdx.x * 256 + threadIdx.x;
    if (i >= total) return;          // total = 3*F*C = 196608
    int c = i & (CC - 1);
    int f = (i >> 8) & (FF - 1);
    int k = i >> 16;                 // F*C == 65536
    Wp[i] = (_Float16)w[((size_t)f * CC + c) * 3 + k];
}

// Zero the two pad rows of an (B,SP,F) f16 activation buffer
__global__ __launch_bounds__(256)
void zero_pads_kernel(_Float16* __restrict__ Hp)
{
    int i = blockIdx.x * 256 + threadIdx.x;   // B*2*F = 8192
    if (i >= BB * 2 * FF) return;
    int f = i & (FF - 1);
    int which = (i >> 8) & 1;
    int b = i >> 9;
    int sp = which ? (SS + 1) : 0;
    Hp[((size_t)(b * SP + sp)) * FF + f] = (_Float16)0.0f;
}

// ---------------------------------------------------------------------------
// K=3 'same' conv as 3 shifted GEMMs, f16 WMMA with f32 accumulation.
// Block = 8 waves; the 96 KB weight slab for this N-quarter (3k x 64f x 256c,
// f16) is staged into LDS once via GLOBAL_LOAD_ASYNC_TO_LDS_B128 (ASYNCcnt),
// then each wave computes one 16(M) x 64(N) tile: 3 shifts * 8 K-steps *
// 4 N-tiles = 96 v_wmma_f32_16x16x32_f16, with B operands from LDS.
//   In : (B, SP, Cin=256) f16 row-major, Wp : (3, F, Cin) f16, bias : (F)
//   Out: Y (B*S, F) f32
// ---------------------------------------------------------------------------
__global__ __launch_bounds__(256)
void conv_gemm_wmma(const _Float16* __restrict__ Xp,
                    const _Float16* __restrict__ Wp,
                    const float*    __restrict__ bias,
                    float* __restrict__ Y)
{
    __shared__ _Float16 Wl[3 * 64 * CC];    // 96 KB (LDS is 320 KB/WGP on CDNA5)

    const int tid = threadIdx.x;
    const int n0  = blockIdx.y * 64;        // N-quarter base column

    // ---- async-copy weight slab: 6144 x 16B chunks, 24 per thread ----
    #pragma unroll
    for (int j = 0; j < 24; ++j) {
        int chunk = j * 256 + tid;          // 16-byte chunk id
        int idx16 = chunk * 8;              // f16 element index in Wl
        int k   = idx16 >> 14;              // / (64*256)
        int rem = idx16 & 16383;
        int fl  = rem >> 8;                 // local column 0..63
        int c   = rem & (CC - 1);
        const _Float16* gsrc = Wp + ((size_t)(k * FF + n0 + fl) * CC + c);
        unsigned lds_addr    = (unsigned)(size_t)(&Wl[idx16]);   // wg-relative LDS byte offset
        unsigned long long ga = (unsigned long long)(size_t)gsrc;
        asm volatile("global_load_async_to_lds_b128 %0, %1, off"
                     :: "v"(lds_addr), "v"(ga) : "memory");
    }
    asm volatile("s_wait_asynccnt 0x0" ::: "memory");
    __syncthreads();

    // ---- per-wave 16x64 tile ----
    const int lane = tid & 31;
    const int wv   = tid >> 5;              // wave 0..7 -> M-tile
    const int lr   = lane & 15;
    const int hi   = (lane >> 4) & 1;       // upper half-wave

    const int m0 = (blockIdx.x * 8 + wv) * 16;
    const int b  = m0 >> 10;
    const int s0 = m0 & (SS - 1);

    v8f acc[4] = {};

    #pragma unroll
    for (int k = 0; k < 3; ++k) {
        // A row for this lane: M = lr, shifted input row s0 + k (pad included)
        const _Float16* ar = Xp + ((size_t)(b * SP + s0 + k) + lr) * CC;
        #pragma unroll
        for (int c0 = 0; c0 < CC; c0 += 32) {
            // A 16x32 f16 layout: lanes 0-15 hold K c0..c0+15,
            // lanes 16-31 hold K c0+16..c0+31 of row M=lr.
            v16h a;
            ((v8h*)&a)[0] = *(const v8h*)(ar + c0 +     (hi ? 16 : 0));
            ((v8h*)&a)[1] = *(const v8h*)(ar + c0 + 8 + (hi ? 16 : 0));
            // B 32x16 f16 layout: lane = column n, lanes 0-15 K=c0..c0+15,
            // lanes 16-31 K=c0+16..c0+31 (16 contiguous halves per lane).
            const int cb = c0 + (hi ? 16 : 0);
            const _Float16* wl = &Wl[((size_t)(k * 64) + lr) * CC + cb];
            v16h b0 = *(const v16h*)(wl);
            v16h b1 = *(const v16h*)(wl + 16 * CC);
            v16h b2 = *(const v16h*)(wl + 32 * CC);
            v16h b3 = *(const v16h*)(wl + 48 * CC);
            acc[0] = __builtin_amdgcn_wmma_f32_16x16x32_f16(false, a, false, b0, (short)0, acc[0], false, false);
            acc[1] = __builtin_amdgcn_wmma_f32_16x16x32_f16(false, a, false, b1, (short)0, acc[1], false, false);
            acc[2] = __builtin_amdgcn_wmma_f32_16x16x32_f16(false, a, false, b2, (short)0, acc[2], false, false);
            acc[3] = __builtin_amdgcn_wmma_f32_16x16x32_f16(false, a, false, b3, (short)0, acc[3], false, false);
        }
    }

    // C/D layout: VGPR v holds row m0 + v + hi*8, column n0 + nt*16 + lr
    #pragma unroll
    for (int nt = 0; nt < 4; ++nt) {
        const int col = n0 + nt * 16 + lr;
        const float bv = bias[col];
        #pragma unroll
        for (int v = 0; v < 8; ++v) {
            const int row = m0 + v + hi * 8;
            Y[(size_t)row * FF + col] = acc[nt][v] + bv;
        }
    }
}

// ---------------------------------------------------------------------------
// LayerNorm(axis=-1) + ReLU over (M, 256), write f16 into padded (B,SP,F)
// ---------------------------------------------------------------------------
__global__ __launch_bounds__(256)
void ln_relu_to_h_kernel(const float* __restrict__ Y,
                         const float* __restrict__ g,
                         const float* __restrict__ bt,
                         _Float16* __restrict__ Hp)
{
    __shared__ float red[256];
    const int r = blockIdx.x;
    const int f = threadIdx.x;
    const int b = r >> 10, s = r & (SS - 1);
    float xv = Y[(size_t)r * FF + f];

    red[f] = xv; __syncthreads();
    #pragma unroll
    for (int off = 128; off > 0; off >>= 1) {
        if (f < off) red[f] += red[f + off];
        __syncthreads();
    }
    float mean = red[0] * (1.0f / 256.0f);
    __syncthreads();
    float d = xv - mean;
    red[f] = d * d; __syncthreads();
    #pragma unroll
    for (int off = 128; off > 0; off >>= 1) {
        if (f < off) red[f] += red[f + off];
        __syncthreads();
    }
    float var = red[0] * (1.0f / 256.0f);
    float val = d * (1.0f / sqrtf(var + 1e-5f)) * g[f] + bt[f];
    val = fmaxf(val, 0.0f);
    Hp[((size_t)(b * SP + s + 1)) * FF + f] = (_Float16)val;
}

// ---------------------------------------------------------------------------
// LayerNorm + ReLU + (F->1 linear) + ReLU fused: reg_len[r]
// ---------------------------------------------------------------------------
__global__ __launch_bounds__(256)
void ln_relu_linear_kernel(const float* __restrict__ Y,
                           const float* __restrict__ g,
                           const float* __restrict__ bt,
                           const float* __restrict__ lw,
                           const float* __restrict__ lb,
                           float* __restrict__ reg)
{
    __shared__ float red[256];
    const int r = blockIdx.x;
    const int f = threadIdx.x;
    float xv = Y[(size_t)r * FF + f];

    red[f] = xv; __syncthreads();
    #pragma unroll
    for (int off = 128; off > 0; off >>= 1) {
        if (f < off) red[f] += red[f + off];
        __syncthreads();
    }
    float mean = red[0] * (1.0f / 256.0f);
    __syncthreads();
    float d = xv - mean;
    red[f] = d * d; __syncthreads();
    #pragma unroll
    for (int off = 128; off > 0; off >>= 1) {
        if (f < off) red[f] += red[f + off];
        __syncthreads();
    }
    float var = red[0] * (1.0f / 256.0f);
    float val = fmaxf(d * (1.0f / sqrtf(var + 1e-5f)) * g[f] + bt[f], 0.0f);
    __syncthreads();
    red[f] = val * lw[f]; __syncthreads();
    #pragma unroll
    for (int off = 128; off > 0; off >>= 1) {
        if (f < off) red[f] += red[f + off];
        __syncthreads();
    }
    if (f == 0) reg[r] = fmaxf(red[0] + lb[0], 0.0f);
}

// ---------------------------------------------------------------------------
// Per-batch inclusive scan of rint(target): fin[b,s] = cumsum(round(target))
// ---------------------------------------------------------------------------
__global__ __launch_bounds__(1024)
void cumsum_round_kernel(const float* __restrict__ target, int* __restrict__ fin)
{
    __shared__ int sh[1024];
    const int b = blockIdx.x, s = threadIdx.x;
    sh[s] = (int)rintf(target[(size_t)b * SS + s]);   // RNE == jnp.round
    __syncthreads();
    for (int off = 1; off < 1024; off <<= 1) {
        int v = (s >= off) ? sh[s - off] : 0;
        __syncthreads();
        sh[s] += v;
        __syncthreads();
    }
    fin[(size_t)b * SS + s] = sh[s];
}

// ---------------------------------------------------------------------------
// expanded[b,t,:] = x[b, s(t), :] where s(t) = first s with fin[s] > t
// (intervals [fin[s]-d[s], fin[s]) are contiguous, so the masked einsum is a
//  gather); zeros when t >= fin[S-1]. One wave per (b,t) row (256 floats).
// ---------------------------------------------------------------------------
__global__ __launch_bounds__(256)
void expand_gather_kernel(const float* __restrict__ x,
                          const int* __restrict__ fin,
                          float* __restrict__ out, int T)
{
    const int wid  = blockIdx.x * 8 + (threadIdx.x >> 5);
    const int lane = threadIdx.x & 31;
    if (wid >= BB * T) return;
    const int b = wid / T, t = wid % T;
    const int* fb = fin + (size_t)b * SS;

    int left = 0, right = SS;                 // lower_bound for fin > t
    while (left < right) {
        int mid = (left + right) >> 1;
        if (fb[mid] > t) right = mid; else left = mid + 1;
    }
    float4 v0 = make_float4(0.f, 0.f, 0.f, 0.f), v1 = v0;
    if (left < SS) {
        const float4* src = (const float4*)(x + ((size_t)(b * SS + left)) * CC);
        v0 = src[lane * 2];
        v1 = src[lane * 2 + 1];
    }
    float4* dst = (float4*)(out + (size_t)wid * CC);
    dst[lane * 2]     = v0;
    dst[lane * 2 + 1] = v1;
}

// ---------------------------------------------------------------------------
extern "C" void kernel_launch(void* const* d_in, const int* in_sizes, int n_in,
                              void* d_out, int out_size, void* d_ws, size_t ws_size,
                              hipStream_t stream)
{
    const float* x   = (const float*)d_in[0];
    const float* tgt = (const float*)d_in[1];
    const float* c1w = (const float*)d_in[2];
    const float* c1b = (const float*)d_in[3];
    const float* c2w = (const float*)d_in[4];
    const float* c2b = (const float*)d_in[5];
    const float* l1g = (const float*)d_in[6];
    const float* l1b = (const float*)d_in[7];
    const float* l2g = (const float*)d_in[8];
    const float* l2b = (const float*)d_in[9];
    const float* lw  = (const float*)d_in[10];
    const float* lb  = (const float*)d_in[11];

    const int M = BB * SS;
    const int T = (out_size - BB * SS) / (BB * CC);   // expanded is (B,T,C)

    float* expanded = (float*)d_out;
    float* reg      = (float*)d_out + (size_t)BB * T * CC;

    // ---- carve workspace (~34 MB) ----
    char*  wsp = (char*)d_ws;
    size_t off = 0;
    auto carve = [&](size_t bytes) -> void* {
        void* p = wsp + off;
        off += (bytes + 255) & ~(size_t)255;
        return p;
    };
    _Float16* Xp  = (_Float16*)carve((size_t)BB * SP * CC * sizeof(_Float16));
    _Float16* Wp1 = (_Float16*)carve((size_t)3 * FF * CC * sizeof(_Float16));
    _Float16* Wp2 = (_Float16*)carve((size_t)3 * FF * CC * sizeof(_Float16));
    float*    Y   = (float*)   carve((size_t)M * FF * sizeof(float));
    _Float16* H1p = (_Float16*)carve((size_t)BB * SP * FF * sizeof(_Float16));
    int*      fin = (int*)     carve((size_t)BB * SS * sizeof(int));

    // ---- pack / convert ----
    {
        int total = BB * SP * CC;
        pack_x_kernel<<<(total + 255) / 256, 256, 0, stream>>>(x, Xp, total);
    }
    {
        int total = 3 * FF * CC;
        pack_w_kernel<<<(total + 255) / 256, 256, 0, stream>>>(c1w, Wp1, total);
        pack_w_kernel<<<(total + 255) / 256, 256, 0, stream>>>(c2w, Wp2, total);
    }
    zero_pads_kernel<<<(BB * 2 * FF + 255) / 256, 256, 0, stream>>>(H1p);

    // ---- duration predictor ----
    dim3 gemm_grid(M / 16 / 8, FF / 64);
    conv_gemm_wmma<<<gemm_grid, 256, 0, stream>>>(Xp, Wp1, c1b, Y);
    ln_relu_to_h_kernel<<<M, 256, 0, stream>>>(Y, l1g, l1b, H1p);
    conv_gemm_wmma<<<gemm_grid, 256, 0, stream>>>(H1p, Wp2, c2b, Y);
    ln_relu_linear_kernel<<<M, 256, 0, stream>>>(Y, l2g, l2b, lw, lb, reg);

    // ---- length regulation (gather) ----
    cumsum_round_kernel<<<BB, 1024, 0, stream>>>(tgt, fin);
    int waves = BB * T;
    expand_gather_kernel<<<(waves + 7) / 8, 256, 0, stream>>>(x, fin, expanded, T);
}
